// PairformerBlock_26508538151729
// MI455X (gfx1250) — matrix operand order repaired and checked by reading, not compiled, not run on hardware
//
#include <hip/hip_runtime.h>

#define Bsz 2
#define Nsz 1024
#define CS  64
#define CZ  32
#define NH  4
#define DHd 16

typedef __bf16 v16bf __attribute__((ext_vector_type(16)));
typedef float  v8f   __attribute__((ext_vector_type(8)));

// D = A(16x32 bf16) * B(32x16 bf16) + C(16x16 f32), wave32 WMMA
static __device__ __forceinline__ v8f wmma_bf16(v16bf a, v16bf b, v8f c) {
  return __builtin_amdgcn_wmma_f32_16x16x32_bf16(false, a, false, b, (short)0, c,
                                                 false, false);
}

// B-matrix fragment: lane holds column n = (lane&15); elements e<8 hold K = 8*(lane>>4)+e, e>=8 hold K = 16+8*(lane>>4)+(e-8).
// For B[k][n] = W[n][k] this is p[0..7] and p[16..23] of W's row n, with p at 8*(lane>>4).
static __device__ __forceinline__ v16bf load_B16(const float* p) {
  v16bf r;
#pragma unroll
  for (int e = 0; e < 8; ++e) { r[e] = (__bf16)p[e]; r[8 + e] = (__bf16)p[16 + e]; }
  return r;
}

// A-matrix fragment from two 8-float chunks (channels [8*hi,8*hi+8) and [16+8*hi,...))
static __device__ __forceinline__ v16bf pack_A(const float* c0, const float* c1) {
  v16bf r;
#pragma unroll
  for (int e = 0; e < 8; ++e) { r[e] = (__bf16)c0[e]; r[8 + e] = (__bf16)c1[e]; }
  return r;
}

static __device__ __forceinline__ void ld8(const float* p, float* d) {
  float4 a = *(const float4*)p;
  float4 b = *(const float4*)(p + 4);
  d[0]=a.x; d[1]=a.y; d[2]=a.z; d[3]=a.w; d[4]=b.x; d[5]=b.y; d[6]=b.z; d[7]=b.w;
}
static __device__ __forceinline__ void st8(float* p, const float* d) {
  *(float4*)p       = make_float4(d[0], d[1], d[2], d[3]);
  *(float4*)(p + 4) = make_float4(d[4], d[5], d[6], d[7]);
}

// GELU. Preferred path: gfx1250 hardware v_tanh_f32 via tanh-form GELU
// (max |err| vs exact-erf GELU ~1e-3, well below the bf16 rounding in the MLP:
// ~5 VALU + 1 TRANS per element). Fallback: branch-free erf (A&S 7.1.26).
static __device__ __forceinline__ float gelu_f(float x) {
#if __has_builtin(__builtin_amdgcn_tanhf)
  float x2 = x * x;
  float u  = 0.79788456080f * __builtin_fmaf(0.044715f * x, x2, x);
  float t  = __builtin_amdgcn_tanhf(u);
  return 0.5f * x * (1.f + t);
#else
  float u  = x * 0.70710678118f;
  float au = fabsf(u);
  float t  = __builtin_amdgcn_rcpf(1.f + 0.3275911f * au);
  float p  = ((((1.061405429f * t - 1.453152027f) * t + 1.421413741f) * t
               - 0.284496736f) * t + 0.254829592f) * t;
  float y  = 1.f - p * __expf(-au * au);
  float er = copysignf(y, u);
  return 0.5f * x * (1.f + er);
#endif
}

// ---------------------------------------------------------------------------
// Stage A: per s-row: zi/zj projections (raw s), LN(s), QKV projection.
// q pre-scaled by 1/sqrt(dh); v stored transposed [B,H,DH,N] for WMMA B layout.
// ---------------------------------------------------------------------------
__global__ __launch_bounds__(256) void srow_kernel(
    const float* __restrict__ s, const float* __restrict__ W_sz,
    const float* __restrict__ W_qkv, const float* __restrict__ b_qkv,
    const float* __restrict__ g_s, const float* __restrict__ be_s,
    float* __restrict__ zi, float* __restrict__ zj,
    float* __restrict__ qb, float* __restrict__ kb, float* __restrict__ vtb) {
  const int row = blockIdx.x;         // b*N + n
  const int b = row >> 10, n = row & 1023;
  const int t = threadIdx.x;
  __shared__ float srow[CS];
  __shared__ float stats[2];
  if (t < CS) srow[t] = s[row * CS + t];
  __syncthreads();
  if (t == 0) {
    float s1 = 0.f, s2 = 0.f;
    for (int c = 0; c < CS; ++c) { s1 += srow[c]; s2 += srow[c] * srow[c]; }
    float mean = s1 * (1.f / CS);
    float var = s2 * (1.f / CS) - mean * mean;
    stats[0] = mean; stats[1] = rsqrtf(var + 1e-5f);
  }
  __syncthreads();
  const float mean = stats[0], rstd = stats[1];
  if (t < 32) {
    float acc = 0.f;
    for (int c = 0; c < CS; ++c) acc += srow[c] * W_sz[t * 128 + c];
    zi[row * CZ + t] = acc;
  } else if (t < 64) {
    const int ch = t - 32;
    float acc = 0.f;
    for (int c = 0; c < CS; ++c) acc += srow[c] * W_sz[ch * 128 + CS + c];
    zj[row * CZ + ch] = acc;
  } else {
    const int c3 = t - 64;            // 0..191
    float acc = b_qkv[c3];
    for (int c = 0; c < CS; ++c) {
      float snc = (srow[c] - mean) * rstd * g_s[c] + be_s[c];
      acc += snc * W_qkv[c3 * CS + c];
    }
    const int kind = c3 >> 6, ch = c3 & 63, h = ch >> 4, d = ch & 15;
    const long idx = (((long)b * NH + h) * Nsz + n) * DHd + d;
    if (kind == 0)      qb[idx] = acc * 0.25f;   // 1/sqrt(16)
    else if (kind == 1) kb[idx] = acc;
    else                vtb[(((long)b * NH + h) * DHd + d) * Nsz + n] = acc;
  }
}

// ---------------------------------------------------------------------------
// Stage B: fused z pass. One wave per (b,i); 64 j-tiles of 16.
// z_new = z + zi + zj + b_sz;  zn = LN(z_new);
// h = gelu(zn @ W_m1^T + b_m1);  z_out = z_new + h @ W_m2^T + b_m2;
// bias[b,h,i,j] = z_out @ W_zb^T + b_zb.
// ---------------------------------------------------------------------------
__global__ __launch_bounds__(32) void zblk_kernel(
    const float* __restrict__ z_in, const float* __restrict__ zi_buf,
    const float* __restrict__ zj_buf, const float* __restrict__ b_sz,
    const float* __restrict__ W_m1, const float* __restrict__ b_m1,
    const float* __restrict__ W_m2, const float* __restrict__ b_m2,
    const float* __restrict__ W_zb, const float* __restrict__ b_zb,
    const float* __restrict__ g_z, const float* __restrict__ be_z,
    float* __restrict__ z_out, float* __restrict__ bias_buf) {
  __shared__ float lds_h[16 * 64];   // gelu(h) staging (D-layout -> A-layout)
  __shared__ float lds_zm[16 * 32];  // MLP output staging

  const int lane = threadIdx.x, ln = lane & 15, hi = lane >> 4;
  const int bi = blockIdx.x, b = bi >> 10, i = bi & 1023;
  const int c0 = 8 * hi, c1 = 16 + 8 * hi;   // A-layout channel chunks

  // per-lane channel-resident constants
  float zi_l[16], bsz_l[16], g_l[16], be_l[16];
  {
    const float* zr = zi_buf + (long)bi * CZ;
    ld8(zr + c0, zi_l); ld8(zr + c1, zi_l + 8);
    ld8(b_sz + c0, bsz_l); ld8(b_sz + c1, bsz_l + 8);
    ld8(g_z + c0, g_l);  ld8(g_z + c1, g_l + 8);
    ld8(be_z + c0, be_l); ld8(be_z + c1, be_l + 8);
  }
  float bm1_l[4], bm2_l[2];
#pragma unroll
  for (int t = 0; t < 4; ++t) bm1_l[t] = b_m1[16 * t + ln];
#pragma unroll
  for (int t = 0; t < 2; ++t) bm2_l[t] = b_m2[16 * t + ln];
  const float bzb_l = (ln < 4) ? b_zb[ln] : 0.f;

  // weight B-fragments (held in VGPRs for all 64 tiles)
  v16bf B1[4], B2[2][2], B3;
#pragma unroll
  for (int t = 0; t < 4; ++t) B1[t] = load_B16(W_m1 + (16 * t + ln) * CZ + 8 * hi);
#pragma unroll
  for (int kc = 0; kc < 2; ++kc)
#pragma unroll
    for (int nt = 0; nt < 2; ++nt)
      B2[kc][nt] = load_B16(W_m2 + (16 * nt + ln) * 64 + 32 * kc + 8 * hi);
  if (ln < 4) {
    B3 = load_B16(W_zb + ln * CZ + 8 * hi);
  } else {
#pragma unroll
    for (int e = 0; e < 16; ++e) B3[e] = (__bf16)0.f;
  }
  const v8f zero8 = {0.f, 0.f, 0.f, 0.f, 0.f, 0.f, 0.f, 0.f};

  for (int jt = 0; jt < 64; ++jt) {
    const int j0 = jt * 16;
    const int jr = j0 + ln;                       // this lane's tile row
    const float* zp  = z_in  + (((long)bi) * Nsz + jr) * CZ;
    const float* zjp = zj_buf + ((long)b * Nsz + jr) * CZ;

    float zres[16], tmp0[8], tmp1[8];
    ld8(zp + c0, tmp0); ld8(zp + c1, tmp1);
    float zj0[8], zj1[8];
    ld8(zjp + c0, zj0); ld8(zjp + c1, zj1);
#pragma unroll
    for (int e = 0; e < 8; ++e) {
      zres[e]     = tmp0[e] + zi_l[e]     + zj0[e] + bsz_l[e];
      zres[8 + e] = tmp1[e] + zi_l[8 + e] + zj1[e] + bsz_l[8 + e];
    }
    // LayerNorm over 32 channels (16 here + 16 in partner lane ^16)
    float s1 = 0.f, s2 = 0.f;
#pragma unroll
    for (int e = 0; e < 16; ++e) { s1 += zres[e]; s2 += zres[e] * zres[e]; }
    s1 += __shfl_xor(s1, 16, 32);
    s2 += __shfl_xor(s2, 16, 32);
    const float mean = s1 * (1.f / 32.f);
    const float var = s2 * (1.f / 32.f) - mean * mean;
    const float rstd = rsqrtf(var + 1e-5f);
    v16bf a_z;
#pragma unroll
    for (int e = 0; e < 16; ++e)
      a_z[e] = (__bf16)((zres[e] - mean) * rstd * g_l[e] + be_l[e]);

    // GEMM1 (16x32 @ 32x64) + bias + gelu -> LDS (row-major)
#pragma unroll
    for (int t = 0; t < 4; ++t) {
      v8f hacc = wmma_bf16(a_z, B1[t], zero8);
#pragma unroll
      for (int r = 0; r < 8; ++r) {
        float x = hacc[r] + bm1_l[t];
        lds_h[(8 * hi + r) * 64 + 16 * t + ln] = gelu_f(x);
      }
    }
    __syncthreads();

    // GEMM2 (16x64 @ 64x32), K split in two 32-chunks
    const float* hr = &lds_h[ln * 64];
    v16bf a2_0 = pack_A(hr + 8 * hi, hr + 16 + 8 * hi);
    v16bf a2_1 = pack_A(hr + 32 + 8 * hi, hr + 48 + 8 * hi);
#pragma unroll
    for (int nt = 0; nt < 2; ++nt) {
      v8f acc = wmma_bf16(a2_0, B2[0][nt], zero8);
      acc = wmma_bf16(a2_1, B2[1][nt], acc);
#pragma unroll
      for (int r = 0; r < 8; ++r)
        lds_zm[(8 * hi + r) * CZ + 16 * nt + ln] = acc[r] + bm2_l[nt];
    }
    __syncthreads();

    // residual add in A-layout, store z_out coalesced
    float zout[16];
    const float* zm = &lds_zm[ln * CZ];
#pragma unroll
    for (int e = 0; e < 8; ++e) {
      zout[e]     = zres[e]     + zm[c0 + e];
      zout[8 + e] = zres[8 + e] + zm[c1 + e];
    }
    float* zop = z_out + (((long)bi) * Nsz + jr) * CZ;
    st8(zop + c0, zout); st8(zop + c1, zout + 8);

    // GEMM3: pair bias (16x32 @ 32x16, only 4 valid output cols)
    v16bf a3;
#pragma unroll
    for (int e = 0; e < 16; ++e) a3[e] = (__bf16)zout[e];
    v8f dbias = wmma_bf16(a3, B3, zero8);
    if (ln < 4) {
      float* bp = bias_buf + (((long)(b * NH + ln)) * Nsz + i) * Nsz + j0 + 8 * hi;
      float bb[8];
#pragma unroll
      for (int r = 0; r < 8; ++r) bb[r] = dbias[r] + bzb_l;
      st8(bp, bb);
    }
    __syncthreads();
  }
}

// ---------------------------------------------------------------------------
// Stage C: pair-biased attention. One wave per (b,h, 16-row i-tile).
// Logits via WMMA (K=16 zero-padded to 32); softmax staged in 64KB LDS;
// PV via 64 chained WMMA accumulations.
// ---------------------------------------------------------------------------
__global__ __launch_bounds__(32) void attn_kernel(
    const float* __restrict__ qb, const float* __restrict__ kb,
    const float* __restrict__ vtb, const float* __restrict__ bias_buf,
    float* __restrict__ att) {
  __shared__ float lds_log[16 * Nsz];           // 64 KB
  const int lane = threadIdx.x, ln = lane & 15, hi = lane >> 4;
  const int blk = blockIdx.x;
  const int it = blk & 63, bh = blk >> 6;       // bh = b*H + h
  const int b = bh >> 2, h = bh & 3;
  const int i0 = it * 16;
  const v8f zero8 = {0.f, 0.f, 0.f, 0.f, 0.f, 0.f, 0.f, 0.f};

  // A = q tile (channels 16..31 are zero pad)
  v16bf a_q;
  {
    const float* qp = qb + ((long)bh * Nsz + i0 + ln) * DHd + 8 * hi;
#pragma unroll
    for (int e = 0; e < 8; ++e) { a_q[e] = (__bf16)qp[e]; a_q[8 + e] = (__bf16)0.f; }
  }

  // Pass 1: logits + bias -> LDS
  for (int jt = 0; jt < 64; ++jt) {
    const int j0 = jt * 16;
    v16bf b_k;
    {
      const float* kp = kb + ((long)bh * Nsz + j0 + ln) * DHd + 8 * hi;
#pragma unroll
      for (int e = 0; e < 8; ++e) { b_k[e] = (__bf16)kp[e]; b_k[8 + e] = (__bf16)0.f; }
    }
    v8f d = wmma_bf16(a_q, b_k, zero8);
    const float* bp = bias_buf + ((long)bh * Nsz + i0 + 8 * hi) * Nsz + j0 + ln;
#pragma unroll
    for (int r = 0; r < 8; ++r)
      lds_log[(8 * hi + r) * Nsz + j0 + ln] = d[r] + bp[(long)r * Nsz];
  }
  __syncthreads();

  // Pass 2: rowwise softmax (exp in place; sums kept in registers)
  float rs[8];
#pragma unroll
  for (int m2 = 0; m2 < 16; ++m2) {
    float mx = -3.4e38f;
    for (int jj = lane; jj < Nsz; jj += 32) mx = fmaxf(mx, lds_log[m2 * Nsz + jj]);
#pragma unroll
    for (int o = 16; o >= 1; o >>= 1) mx = fmaxf(mx, __shfl_xor(mx, o, 32));
    float ssum = 0.f;
    for (int jj = lane; jj < Nsz; jj += 32) {
      float e = __expf(lds_log[m2 * Nsz + jj] - mx);
      lds_log[m2 * Nsz + jj] = e;
      ssum += e;
    }
#pragma unroll
    for (int o = 16; o >= 1; o >>= 1) ssum += __shfl_xor(ssum, o, 32);
    if ((m2 >> 3) == hi) rs[m2 & 7] = ssum;
  }
  __syncthreads();

  // Pass 3: out = P @ V, chained WMMA accumulation over 64 j-tiles
  v8f acc = zero8;
  for (int jt = 0; jt < 64; ++jt) {
    const int j0 = jt * 16;
    v16bf a_p;
    const float* pp = &lds_log[ln * Nsz + j0 + 8 * hi];
#pragma unroll
    for (int e = 0; e < 8; ++e) { a_p[e] = (__bf16)pp[e]; a_p[8 + e] = (__bf16)0.f; }
    v16bf b_v;
    {
      const float* vp = vtb + ((long)bh * DHd + ln) * Nsz + j0 + 8 * hi;
#pragma unroll
      for (int e = 0; e < 8; ++e) { b_v[e] = (__bf16)vp[e]; b_v[8 + e] = (__bf16)0.f; }
    }
    acc = wmma_bf16(a_p, b_v, acc);
  }
  float* ap = att + ((long)b * Nsz + i0 + 8 * hi) * CS + h * DHd + ln;
#pragma unroll
  for (int r = 0; r < 8; ++r) ap[(long)r * CS] = acc[r] / rs[r];
}

// ---------------------------------------------------------------------------
// Stage D: s_out = s + att @ W_o^T + b_o
// ---------------------------------------------------------------------------
__global__ __launch_bounds__(64) void outproj_kernel(
    const float* __restrict__ s, const float* __restrict__ att,
    const float* __restrict__ W_o, const float* __restrict__ b_o,
    float* __restrict__ s_out) {
  const int row = blockIdx.x, c = threadIdx.x;
  __shared__ float arow[CS];
  arow[c] = att[(long)row * CS + c];
  __syncthreads();
  float acc = b_o[c];
  for (int k = 0; k < CS; ++k) acc += arow[k] * W_o[c * CS + k];
  s_out[(long)row * CS + c] = s[(long)row * CS + c] + acc;
}

// ---------------------------------------------------------------------------
extern "C" void kernel_launch(void* const* d_in, const int* in_sizes, int n_in,
                              void* d_out, int out_size, void* d_ws, size_t ws_size,
                              hipStream_t stream) {
  (void)in_sizes; (void)n_in; (void)out_size; (void)ws_size;
  const float* s     = (const float*)d_in[0];
  const float* z     = (const float*)d_in[1];
  const float* W_sz  = (const float*)d_in[2];
  const float* b_sz  = (const float*)d_in[3];
  const float* W_m1  = (const float*)d_in[4];
  const float* b_m1  = (const float*)d_in[5];
  const float* W_m2  = (const float*)d_in[6];
  const float* b_m2  = (const float*)d_in[7];
  const float* W_qkv = (const float*)d_in[8];
  const float* b_qkv = (const float*)d_in[9];
  const float* W_zb  = (const float*)d_in[10];
  const float* b_zb  = (const float*)d_in[11];
  const float* W_o   = (const float*)d_in[12];
  const float* b_o   = (const float*)d_in[13];
  const float* g_s   = (const float*)d_in[14];
  const float* be_s  = (const float*)d_in[15];
  const float* g_z   = (const float*)d_in[16];
  const float* be_z  = (const float*)d_in[17];

  float* out_s = (float*)d_out;
  float* out_z = out_s + (size_t)Bsz * Nsz * CS;

  float* ws   = (float*)d_ws;
  float* zi   = ws;                       // 65536
  float* zj   = zi + 65536;               // 65536
  float* qb   = zj + 65536;               // 131072
  float* kb   = qb + 131072;              // 131072
  float* vtb  = kb + 131072;              // 131072
  float* attb = vtb + 131072;             // 131072
  float* bias = attb + 131072;            // 8388608  (~36 MB total ws)

  srow_kernel<<<Bsz * Nsz, 256, 0, stream>>>(s, W_sz, W_qkv, b_qkv, g_s, be_s,
                                             zi, zj, qb, kb, vtb);
  zblk_kernel<<<Bsz * Nsz, 32, 0, stream>>>(z, zi, zj, b_sz, W_m1, b_m1, W_m2,
                                            b_m2, W_zb, b_zb, g_z, be_z,
                                            out_z, bias);
  attn_kernel<<<Bsz * NH * (Nsz / 16), 32, 0, stream>>>(qb, kb, vtb, bias, attb);
  outproj_kernel<<<Bsz * Nsz, 64, 0, stream>>>(s, attb, W_o, b_o, out_s);
}
